// Attention_3307124818205
// MI455X (gfx1250) — matrix operand order repaired and checked by reading, not compiled, not run on hardware
//
#include <hip/hip_runtime.h>

typedef __bf16 bf16_t;
typedef __attribute__((ext_vector_type(16))) __bf16 v16bf;
typedef __attribute__((ext_vector_type(8)))  float  v8f;
typedef __attribute__((ext_vector_type(4)))  int    v4i_t;

#define NHEADS 12
#define HD     64
#define SEQ    1024
#define BATCH  8
#define CDIM   768
#define EPSF   1e-6f

// ------------------------------------------------- async global->LDS path ---

#if defined(__AMDGCN__) && __has_builtin(__builtin_amdgcn_global_load_async_to_lds_b128)
#define HAVE_ASYNC_LDS 1
#else
#define HAVE_ASYNC_LDS 0
#endif

// Copy 16 bytes global -> LDS. Async (no VGPR staging) when available.
static __device__ inline void async_cp16(const bf16_t* g, bf16_t* l) {
#if HAVE_ASYNC_LDS
  __builtin_amdgcn_global_load_async_to_lds_b128((v4i_t*)g, (v4i_t*)l, 0, 0);
#else
  *(uint4*)l = *(const uint4*)g;
#endif
}

static __device__ inline void wait_async() {
#if HAVE_ASYNC_LDS
#if __has_builtin(__builtin_amdgcn_s_wait_asynccnt)
  __builtin_amdgcn_s_wait_asynccnt(0);
#else
  asm volatile("s_wait_asynccnt 0x0" ::: "memory");
#endif
#endif
}

// ---------------------------------------------------------------- helpers ---

static __device__ inline v8f wmma_bf16(v16bf a, v16bf b, v8f c) {
  // D = A(16x32) * B(32x16) + C, f32 accumulate
  return __builtin_amdgcn_wmma_f32_16x16x32_bf16(false, a, false, b, (short)0, c,
                                                 false, false);
}

// Assemble a 16-element bf16 fragment from two 16-byte chunks (LDS or global).
static __device__ inline v16bf load_frag2(const bf16_t* p0, const bf16_t* p1) {
  union { uint4 q[2]; v16bf v; } u;
  u.q[0] = *(const uint4*)p0;
  u.q[1] = *(const uint4*)p1;
  return u.v;
}

// ------------------------------------------------------------- converters ---

// x (N,B,C) f32  ->  xb (B*N, C) bf16   (row r = b*N + n)
__global__ __launch_bounds__(256) void cvt_x_kernel(const float* __restrict__ x,
                                                    bf16_t* __restrict__ xb) {
  long idx = (long)blockIdx.x * 256 + threadIdx.x;
  if (idx >= (long)SEQ * BATCH * CDIM) return;
  int c = (int)(idx % CDIM);
  long t = idx / CDIM;
  int b = (int)(t % BATCH);
  long n = t / BATCH;
  xb[((long)b * SEQ + n) * CDIM + c] = (bf16_t)x[idx];
}

// w (R, Ccols) f32 -> wt (Ccols, R) bf16
__global__ __launch_bounds__(256) void cvt_tr_kernel(const float* __restrict__ w,
                                                     bf16_t* __restrict__ wt,
                                                     int R, int Ccols) {
  long idx = (long)blockIdx.x * 256 + threadIdx.x;
  if (idx >= (long)R * Ccols) return;
  int col = (int)(idx % Ccols);
  int row = (int)(idx / Ccols);
  wt[(long)col * R + row] = (bf16_t)w[idx];
}

// ------------------------------------------------------------ GEMM common ---
// 128(M) x 64(N) block tile, 8 waves, each 32x32 = 2x2 WMMA fragments.
// Double-buffered LDS; async global->LDS copies overlap the WMMA main loop.

#define LDA_S 72  // padded LDS strides (bf16 elems); 72*2=144B keeps 16B align
#define LDB_S 72

static __device__ inline void gemm_mainloop(const bf16_t* __restrict__ A, int lda,
                                            const bf16_t* __restrict__ Bt, int ldb,
                                            int K, bf16_t (*As)[128 * LDA_S],
                                            bf16_t (*Bs)[64 * LDB_S], v8f c[2][2]) {
  const int tid = threadIdx.x;
  const int w = tid >> 5, lane = tid & 31;
  const int wm = w & 3, wn = w >> 2;
  const int l15 = lane & 15, half = lane >> 4;

  auto stage = [&](int kc, int buf) {
#pragma unroll
    for (int i = 0; i < 4; ++i) {  // A tile: 128 x 64
      int ch = tid + i * 256;
      int r = ch >> 3, c8 = ch & 7;
      async_cp16(&A[(size_t)r * lda + kc + c8 * 8], &As[buf][r * LDA_S + c8 * 8]);
    }
#pragma unroll
    for (int i = 0; i < 2; ++i) {  // B^T tile: 64(N) x 64(K)
      int ch = tid + i * 256;
      int r = ch >> 3, c8 = ch & 7;
      async_cp16(&Bt[(size_t)r * ldb + kc + c8 * 8], &Bs[buf][r * LDB_S + c8 * 8]);
    }
  };

  stage(0, 0);
  wait_async();
  __syncthreads();

  for (int kc = 0; kc < K; kc += 64) {
    const int p = (kc >> 6) & 1;
    if (kc + 64 < K) stage(kc + 64, p ^ 1);  // overlap next-chunk DMA with WMMAs
#pragma unroll
    for (int kk = 0; kk < 64; kk += 32) {
      v16bf af[2], bb[2];
#pragma unroll
      for (int mf = 0; mf < 2; ++mf) {
        const bf16_t* pa = &As[p][(wm * 32 + mf * 16 + l15) * LDA_S + kk + half * 8];
        af[mf] = load_frag2(pa, pa + 16);
      }
#pragma unroll
      for (int nf = 0; nf < 2; ++nf) {
        const bf16_t* pb = &Bs[p][(wn * 32 + nf * 16 + l15) * LDB_S + kk + half * 8];
        bb[nf] = load_frag2(pb, pb + 16);
      }
#pragma unroll
      for (int mf = 0; mf < 2; ++mf)
#pragma unroll
        for (int nf = 0; nf < 2; ++nf)
          c[mf][nf] = wmma_bf16(af[mf], bb[nf], c[mf][nf]);
    }
    wait_async();
    __syncthreads();
  }
}

// ------------------------------------------------------------- QKV kernel ---
// xb (B*N, C) @ Wqkvt^T + b_qkv -> Q,K (B,H,N,hd) bf16 ; V transposed (B,H,hd,N)

__global__ __launch_bounds__(256) void qkv_gemm_kernel(
    const bf16_t* __restrict__ xb, const bf16_t* __restrict__ Wt,
    const float* __restrict__ bias, bf16_t* __restrict__ Q,
    bf16_t* __restrict__ Kb, bf16_t* __restrict__ Vt) {
  __shared__ alignas(16) bf16_t As[2][128 * LDA_S];
  __shared__ alignas(16) bf16_t Bs[2][64 * LDB_S];
  const int mt = blockIdx.x;  // 64 row tiles
  const int nt = blockIdx.y;  // 36 col tiles (3*C / 64)
  v8f c[2][2] = {};
  gemm_mainloop(xb + (size_t)mt * 128 * CDIM, CDIM,
                Wt + (size_t)nt * 64 * CDIM, CDIM, CDIM, As, Bs, c);
  const int tid = threadIdx.x;
  const int w = tid >> 5, lane = tid & 31;
  const int wm = w & 3, wn = w >> 2;
  const int l15 = lane & 15, half = lane >> 4;
#pragma unroll
  for (int mf = 0; mf < 2; ++mf)
#pragma unroll
    for (int nf = 0; nf < 2; ++nf) {
      int j = nt * 64 + wn * 32 + nf * 16 + l15;  // 0..2303
      float bj = bias[j];
      int part = j / CDIM;
      int jr = j % CDIM;
      int hh = jr >> 6, d = jr & 63;
      int row0 = mt * 128 + wm * 32 + mf * 16 + half * 8;  // r = b*N + n
      int b = row0 >> 10, n0 = row0 & 1023;
      int bh = b * NHEADS + hh;
      if (part == 2) {
        // V transposed: lane's 8 values are consecutive tokens of one feature
        union { uint4 q; bf16_t e[8]; } u;
#pragma unroll
        for (int v = 0; v < 8; ++v) u.e[v] = (bf16_t)(c[mf][nf][v] + bj);
        *(uint4*)&Vt[((size_t)bh * HD + d) * SEQ + n0] = u.q;
      } else {
        bf16_t* dst = (part == 0) ? Q : Kb;
#pragma unroll
        for (int v = 0; v < 8; ++v)
          dst[((size_t)bh * SEQ + n0 + v) * HD + d] = (bf16_t)(c[mf][nf][v] + bj);
      }
    }
}

// -------------------------------------------------- flash attention kernel --
// grid: (B*H, N/128). 256 threads = 8 waves; wave w owns 16 query rows.
// Q fragments register-resident; K/V^T tiles double-buffered via async DMA.

__global__ __launch_bounds__(256) void attn_kernel(
    const bf16_t* __restrict__ Q, const bf16_t* __restrict__ K,
    const bf16_t* __restrict__ Vt, const float* __restrict__ policy,
    bf16_t* __restrict__ ctx) {
  __shared__ alignas(16) bf16_t Ks[2][64 * 72];
  __shared__ alignas(16) bf16_t Vts[2][64 * 72];  // V^T tile: [d][m]
  __shared__ alignas(16) bf16_t Ps[8][16 * 72];   // per-wave P staging
  __shared__ float pols[2][64];
  __shared__ float vsumLds[4][64];

  const int bh = blockIdx.x;
  const int b = bh / NHEADS, h = bh % NHEADS;
  const int qt = blockIdx.y;
  const int tid = threadIdx.x;
  const int w = tid >> 5, lane = tid & 31;
  const int l15 = lane & 15, half = lane >> 4;
  const size_t baseK = (size_t)bh * SEQ * HD;  // K: (bh, n, d)
  const size_t baseV = (size_t)bh * HD * SEQ;  // Vt: (bh, d, n)
  const float scale = 0.125f;                  // hd^-0.5

  // Q A-fragments are loop-invariant: load straight to registers.
  v16bf qa[2];
  {
    const bf16_t* Qg = Q + baseK + (size_t)(qt * 128 + w * 16 + l15) * HD;
#pragma unroll
    for (int kk = 0; kk < 2; ++kk) {
      const bf16_t* p = Qg + kk * 32 + half * 8;
      qa[kk] = load_frag2(p, p + 16);
    }
  }

  v8f o[4] = {};
  float rmax[8], rden[8];
#pragma unroll
  for (int v = 0; v < 8; ++v) { rmax[v] = -1e30f; rden[v] = 0.0f; }
  float vsacc = 0.0f;
  const int vd = tid & 63, vg = tid >> 6;

  auto load_tiles = [&](int mt, int buf) {
    const bf16_t* Kg = K + baseK + (size_t)mt * 64 * HD;
#pragma unroll
    for (int i = 0; i < 2; ++i) {  // K tile 64(m) x 64(d), row-major
      int ch = tid + i * 256;
      int r = ch >> 3, c8 = ch & 7;
      async_cp16(Kg + (size_t)r * HD + c8 * 8, &Ks[buf][r * 72 + c8 * 8]);
    }
    const bf16_t* Vg = Vt + baseV + (size_t)mt * 64;  // rows d, stride SEQ
#pragma unroll
    for (int i = 0; i < 2; ++i) {  // V^T tile 64(d) x 64(m)
      int ch = tid + i * 256;
      int r = ch >> 3, c8 = ch & 7;
      async_cp16(Vg + (size_t)r * SEQ + c8 * 8, &Vts[buf][r * 72 + c8 * 8]);
    }
    if (tid < 64) pols[buf][tid] = policy[(size_t)b * SEQ + mt * 64 + tid];
  };

  load_tiles(0, 0);
  wait_async();
  __syncthreads();

  for (int mt = 0; mt < SEQ / 64; ++mt) {
    const int p = mt & 1;
    if (mt + 1 < SEQ / 64) load_tiles(mt + 1, p ^ 1);  // overlap with compute

    // V column-sum accumulation (for the +eps/N numerator term)
    {
      float s = 0.0f;
      for (int m = vg * 16; m < vg * 16 + 16; ++m) s += (float)Vts[p][vd * 72 + m];
      vsacc += s;
    }

    // S = Q K^T (wave rows: qt*128 + w*16 .. +15; cols: 64 keys of tile)
    v8f sfr[4] = {};
#pragma unroll
    for (int kk = 0; kk < 64; kk += 32) {
#pragma unroll
      for (int nf = 0; nf < 4; ++nf) {
        const bf16_t* pb = &Ks[p][(nf * 16 + l15) * 72 + kk + half * 8];
        v16bf kf = load_frag2(pb, pb + 16);
        sfr[nf] = wmma_bf16(qa[kk >> 5], kf, sfr[nf]);
      }
    }

    // ---- online softmax (rows live across 16-lane halves of the wave)
#pragma unroll
    for (int v = 0; v < 8; ++v) {
      float tmax = -1e30f;
#pragma unroll
      for (int nf = 0; nf < 4; ++nf) tmax = fmaxf(tmax, sfr[nf][v] * scale);
      for (int off = 1; off < 16; off <<= 1)
        tmax = fmaxf(tmax, __shfl_xor(tmax, off, 32));
      float nm = fmaxf(rmax[v], tmax);
      float corr = __expf(rmax[v] - nm);
      rmax[v] = nm;
      rden[v] *= corr;
#pragma unroll
      for (int df = 0; df < 4; ++df) o[df][v] *= corr;

      int n_global = qt * 128 + w * 16 + half * 8 + v;
      float psum = 0.0f;
#pragma unroll
      for (int nf = 0; nf < 4; ++nf) {
        int m_local = nf * 16 + l15;
        int m_global = mt * 64 + m_local;
        float ap = pols[p][m_local];
        if (m_global == n_global) ap = 1.0f;  // ap + (1-ap)*eye
        float pe = __expf(sfr[nf][v] * scale - nm) * ap;
        psum += pe;
        Ps[w][(half * 8 + v) * 72 + m_local] = (bf16_t)pe;
      }
      for (int off = 1; off < 16; off <<= 1) psum += __shfl_xor(psum, off, 32);
      rden[v] += psum;
    }
    // Ps slice is wave-private and LDS ops are in-order per wave: no barrier.

    // O += P (16x64) * V (64x64)
#pragma unroll
    for (int kk = 0; kk < 64; kk += 32) {
      const bf16_t* pp = &Ps[w][l15 * 72 + kk + half * 8];
      v16bf pa = load_frag2(pp, pp + 16);
#pragma unroll
      for (int df = 0; df < 4; ++df) {
        const bf16_t* pv = &Vts[p][(df * 16 + l15) * 72 + kk + half * 8];
        v16bf vb = load_frag2(pv, pv + 16);
        o[df] = wmma_bf16(pa, vb, o[df]);
      }
    }
    wait_async();
    __syncthreads();
  }

  vsumLds[vg][vd] = vsacc;
  __syncthreads();

  // out = (O + (eps/N)*colsum(V)) / (den + eps) -> ctx[(n*B+b)][h*64+d] bf16
  const float epsn = EPSF / (float)SEQ;
#pragma unroll
  for (int df = 0; df < 4; ++df) {
    int d = df * 16 + l15;
    float vs = vsumLds[0][d] + vsumLds[1][d] + vsumLds[2][d] + vsumLds[3][d];
#pragma unroll
    for (int v = 0; v < 8; ++v) {
      int n = qt * 128 + w * 16 + half * 8 + v;
      float val = (o[df][v] + epsn * vs) / (rden[v] + EPSF);
      ctx[((size_t)n * BATCH + b) * CDIM + h * 64 + d] = (bf16_t)val;
    }
  }
}

// ------------------------------------------------------------- proj kernel --
// ctx (rows n*B+b, C) @ Wpt^T + b_proj -> out f32 (N,B,C)

__global__ __launch_bounds__(256) void proj_gemm_kernel(
    const bf16_t* __restrict__ ctx, const bf16_t* __restrict__ Wpt,
    const float* __restrict__ bias, float* __restrict__ out) {
  __shared__ alignas(16) bf16_t As[2][128 * LDA_S];
  __shared__ alignas(16) bf16_t Bs[2][64 * LDB_S];
  const int mt = blockIdx.x;  // 64
  const int nt = blockIdx.y;  // 12
  v8f c[2][2] = {};
  gemm_mainloop(ctx + (size_t)mt * 128 * CDIM, CDIM,
                Wpt + (size_t)nt * 64 * CDIM, CDIM, CDIM, As, Bs, c);
  const int tid = threadIdx.x;
  const int w = tid >> 5, lane = tid & 31;
  const int wm = w & 3, wn = w >> 2;
  const int l15 = lane & 15, half = lane >> 4;
#pragma unroll
  for (int mf = 0; mf < 2; ++mf)
#pragma unroll
    for (int nf = 0; nf < 2; ++nf) {
      int j = nt * 64 + wn * 32 + nf * 16 + l15;
      float bj = bias[j];
#pragma unroll
      for (int v = 0; v < 8; ++v) {
        int row = mt * 128 + wm * 32 + mf * 16 + half * 8 + v;
        out[(size_t)row * CDIM + j] = c[mf][nf][v] + bj;
      }
    }
}

// ---------------------------------------------------------------- launcher --

extern "C" void kernel_launch(void* const* d_in, const int* in_sizes, int n_in,
                              void* d_out, int out_size, void* d_ws, size_t ws_size,
                              hipStream_t stream) {
  (void)in_sizes; (void)n_in; (void)out_size; (void)ws_size;
  const float* x      = (const float*)d_in[0];
  const float* policy = (const float*)d_in[1];
  const float* W_qkv  = (const float*)d_in[2];
  const float* b_qkv  = (const float*)d_in[3];
  const float* W_proj = (const float*)d_in[4];
  const float* b_proj = (const float*)d_in[5];
  float* out = (float*)d_out;

  char* ws = (char*)d_ws;
  size_t off = 0;
  auto alloc = [&](size_t bytes) {
    void* p = ws + off;
    off += (bytes + 255) & ~(size_t)255;
    return p;
  };
  const size_t NTOK = (size_t)SEQ * BATCH;  // 8192
  bf16_t* xb    = (bf16_t*)alloc(NTOK * CDIM * 2);
  bf16_t* Wqkvt = (bf16_t*)alloc((size_t)3 * CDIM * CDIM * 2);
  bf16_t* Wpt   = (bf16_t*)alloc((size_t)CDIM * CDIM * 2);
  bf16_t* Qb    = (bf16_t*)alloc(NTOK * CDIM * 2);
  bf16_t* Kb    = (bf16_t*)alloc(NTOK * CDIM * 2);
  bf16_t* Vtb   = (bf16_t*)alloc(NTOK * CDIM * 2);  // (B,H,hd,N)
  bf16_t* ctx   = (bf16_t*)alloc(NTOK * CDIM * 2);

  {
    long total = (long)NTOK * CDIM;
    cvt_x_kernel<<<(unsigned)((total + 255) / 256), 256, 0, stream>>>(x, xb);
  }
  {
    long total = (long)CDIM * 3 * CDIM;
    cvt_tr_kernel<<<(unsigned)((total + 255) / 256), 256, 0, stream>>>(
        W_qkv, Wqkvt, CDIM, 3 * CDIM);
  }
  {
    long total = (long)CDIM * CDIM;
    cvt_tr_kernel<<<(unsigned)((total + 255) / 256), 256, 0, stream>>>(
        W_proj, Wpt, CDIM, CDIM);
  }
  qkv_gemm_kernel<<<dim3(64, 36), 256, 0, stream>>>(xb, Wqkvt, b_qkv, Qb, Kb, Vtb);
  attn_kernel<<<dim3(BATCH * NHEADS, SEQ / 128), 256, 0, stream>>>(
      Qb, Kb, Vtb, policy, ctx);
  proj_gemm_kernel<<<dim3(64, 12), 256, 0, stream>>>(ctx, Wpt, b_proj, out);
}
